// GraphRec_11304353923460
// MI455X (gfx1250) — matrix operand order repaired and checked by reading, not compiled, not run on hardware
//
#include <hip/hip_runtime.h>

#define L_LEN 50

typedef __attribute__((ext_vector_type(16))) _Float16 v16h;
typedef __attribute__((ext_vector_type(8)))  _Float16 v8h;
typedef __attribute__((ext_vector_type(8)))  float    v8f;

__device__ __forceinline__ v16h cat8(v8h lo, v8h hi) {
  return __builtin_shufflevector(lo, hi, 0,1,2,3,4,5,6,7,8,9,10,11,12,13,14,15);
}

// ---------------------------------------------------------------------------
// Wave-level GEMM: Y[16,64] = relu(X[16,K] @ W[K,64] + bias)
// X in LDS f16 row-major (pitch xPitch halfs), WT in LDS f16 transposed [64][K].
// Each wave handles rows [rowBase, rowBase+16). Uses v_wmma_f32_16x16x32_f16.
// A frag: lane row = rowBase+(lane&15); halves {k..k+7, k+16..k+23} per lane half.
// B frag: col n = nt*16+(lane&15); halves {k..k+7, k+16..k+23} per lane half.
// C frag: row = rowBase + v + 8*(lane>=16), col = nt*16+(lane&15).
// ---------------------------------------------------------------------------
template<int K>
__device__ __forceinline__ void waveGemm(const _Float16* X, int xPitch,
                                         const _Float16* WT,
                                         const float* bias,
                                         _Float16* outH, int outPitch,
                                         float* outF, int outFPitch,
                                         int rowBase, int lane)
{
  const int rloc = lane & 15;
  const int hs   = lane >> 4;          // which half of the wave
  const int row  = rowBase + rloc;
  constexpr int NK = K / 32;

  v16h afrag[NK];
  const _Float16* xr = X + row * xPitch + hs * 8;
#pragma unroll
  for (int kc = 0; kc < NK; ++kc) {
    v8h a0 = *(const v8h*)(xr + kc * 32);
    v8h a1 = *(const v8h*)(xr + kc * 32 + 16);
    afrag[kc] = cat8(a0, a1);
  }

#pragma unroll
  for (int nt = 0; nt < 4; ++nt) {
    const int col = nt * 16 + rloc;
    const _Float16* wr = WT + col * K + hs * 8;
    v8f acc = {0.f, 0.f, 0.f, 0.f, 0.f, 0.f, 0.f, 0.f};
#pragma unroll
    for (int kc = 0; kc < NK; ++kc) {
      v8h b0 = *(const v8h*)(wr + kc * 32);
      v8h b1 = *(const v8h*)(wr + kc * 32 + 16);
      v16h bfrag = cat8(b0, b1);
      acc = __builtin_amdgcn_wmma_f32_16x16x32_f16(false, afrag[kc], false, bfrag,
                                                   (short)0, acc, false, false);
    }
    const float bv = bias[col];
#pragma unroll
    for (int v = 0; v < 8; ++v) {
      const int r = rowBase + v + hs * 8;
      float val = acc[v] + bv;
      val = fmaxf(val, 0.0f);
      if (outH) outH[r * outPitch + col] = (_Float16)val;
      if (outF) outF[(size_t)r * outFPitch + col] = val;
    }
  }
}

// W global [K][64] row-major -> dst LDS transposed [64][K] f16
__device__ __forceinline__ void loadWT(const float* __restrict__ W, int K,
                                       _Float16* dst, int tid)
{
  for (int i = tid; i < K * 64; i += 128) {
    int k = i >> 6, n = i & 63;
    dst[n * K + k] = (_Float16)W[i];
  }
}

// ---------------------------------------------------------------------------
// Branch kernel: per batch row b, computes
//   feat = hasMlp ? relu-MLP(concat(tblN[idxN], tblR[idxR])) : tblN[idxN]   [L,64]
//   a    = softmax(attMLP(concat(feat, u_rep)))                              [L]
//   out  = sum_l a[l] * feat[l]                                              [64]
// 128 threads (4 wave32), one 16-row WMMA tile per wave.
// ---------------------------------------------------------------------------
struct BranchSmem {
  _Float16 wT[64 * 128];    // current-stage transposed weights (refilled per layer)
  _Float16 bufA[64 * 128];  // MLP concat input; reused as att layer-2 output
  _Float16 bufB[64 * 128];  // att concat input: [feat | u_rep]
  _Float16 bufC[64 * 64];   // hidden
  float biasM1[64], biasM2[64], biasA1[64], biasA2[64];
  float w3[64];
  float urep[64];
  float logits[64];
  float alpha[64];
  int idxN[64], idxR[64];
};

__global__ void __launch_bounds__(128)
branch_kernel(const int* __restrict__ idxN, const int* __restrict__ idxR,
              const float* __restrict__ tblN, const float* __restrict__ tblR,
              const int* __restrict__ uIdx, const float* __restrict__ uTbl,
              const float* mW1, const float* mB1, const float* mW2, const float* mB2,
              const float* aW1, const float* aB1, const float* aW2, const float* aB2,
              const float* aW3, const float* aB3,
              float* __restrict__ outVec)
{
  __shared__ BranchSmem sm;
  const int tid = threadIdx.x;
  const int lane = tid & 31;
  const int rowBase = (tid >> 5) * 16;
  const int b = blockIdx.x;
  const bool hasMlp = (mW1 != nullptr);

  if (tid < 64) {
    sm.urep[tid]  = uTbl[(size_t)uIdx[b] * 64 + tid];
    sm.biasA1[tid] = aB1[tid];
    sm.biasA2[tid] = aB2[tid];
    sm.w3[tid]     = aW3[tid];
    if (hasMlp) { sm.biasM1[tid] = mB1[tid]; sm.biasM2[tid] = mB2[tid]; }
    sm.idxN[tid] = (tid < L_LEN) ? idxN[(size_t)b * L_LEN + tid] : 0;
    sm.idxR[tid] = (hasMlp && tid < L_LEN) ? idxR[(size_t)b * L_LEN + tid] : 0;
  }
  __syncthreads();

  if (hasMlp) {
    for (int i = tid; i < 64 * 128; i += 128) {   // gather concat(neigh, rating)
      int row = i >> 7, c = i & 127;
      float v = 0.f;
      if (row < L_LEN)
        v = (c < 64) ? tblN[(size_t)sm.idxN[row] * 64 + c]
                     : tblR[(size_t)sm.idxR[row] * 64 + (c - 64)];
      sm.bufA[i] = (_Float16)v;
    }
    for (int i = tid; i < 64 * 64; i += 128) {    // broadcast u_rep into bufB[:,64:]
      int row = i >> 6, c = i & 63;
      sm.bufB[row * 128 + 64 + c] = (_Float16)sm.urep[c];
    }
    loadWT(mW1, 128, sm.wT, tid);
    __syncthreads();
    waveGemm<128>(sm.bufA, 128, sm.wT, sm.biasM1, sm.bufC, 64, nullptr, 0, rowBase, lane);
    __syncthreads();
    loadWT(mW2, 64, sm.wT, tid);
    __syncthreads();
    waveGemm<64>(sm.bufC, 64, sm.wT, sm.biasM2, sm.bufB, 128, nullptr, 0, rowBase, lane);
    __syncthreads();
  } else {
    for (int i = tid; i < 64 * 64; i += 128) {    // feat = raw embeddings
      int row = i >> 6, c = i & 63;
      float v = (row < L_LEN) ? tblN[(size_t)sm.idxN[row] * 64 + c] : 0.f;
      sm.bufB[row * 128 + c]      = (_Float16)v;
      sm.bufB[row * 128 + 64 + c] = (_Float16)sm.urep[c];
    }
    __syncthreads();
  }

  loadWT(aW1, 128, sm.wT, tid);
  __syncthreads();
  waveGemm<128>(sm.bufB, 128, sm.wT, sm.biasA1, sm.bufC, 64, nullptr, 0, rowBase, lane);
  __syncthreads();
  loadWT(aW2, 64, sm.wT, tid);
  __syncthreads();
  waveGemm<64>(sm.bufC, 64, sm.wT, sm.biasA2, sm.bufA, 64, nullptr, 0, rowBase, lane);
  __syncthreads();

  if (tid < 64) {                                  // logits = h @ w3 + b3
    float s = aB3[0];
    for (int c = 0; c < 64; ++c) s += (float)sm.bufA[tid * 64 + c] * sm.w3[c];
    sm.logits[tid] = s;
  }
  __syncthreads();
  if (tid == 0) {                                  // softmax over L valid rows
    float mx = -1e30f;
    for (int l = 0; l < L_LEN; ++l) mx = fmaxf(mx, sm.logits[l]);
    float se = 0.f;
    for (int l = 0; l < L_LEN; ++l) { float e = expf(sm.logits[l] - mx); sm.alpha[l] = e; se += e; }
    float inv = 1.f / se;
    for (int l = 0; l < L_LEN; ++l) sm.alpha[l] *= inv;
  }
  __syncthreads();
  if (tid < 64) {                                  // weighted sum of features
    float s = 0.f;
    for (int l = 0; l < L_LEN; ++l) s += sm.alpha[l] * (float)sm.bufB[l * 128 + tid];
    outVec[(size_t)b * 64 + tid] = s;
  }
}

// ---------------------------------------------------------------------------
// umlp: out[B,64] = relu(relu(concat(X0,X1)[B,128] @ w1 + b1) @ w2 + b2)
// ---------------------------------------------------------------------------
struct Mlp2Smem {
  _Float16 w1T[64 * 128];
  _Float16 w2T[64 * 64];
  _Float16 bufX[64 * 128];
  _Float16 bufH[64 * 64];
  float b1s[64], b2s[64];
};

__global__ void __launch_bounds__(128)
mlp2_kernel(const float* __restrict__ X0, const float* __restrict__ X1,
            const float* __restrict__ w1, const float* __restrict__ b1,
            const float* __restrict__ w2, const float* __restrict__ b2,
            float* __restrict__ out)
{
  __shared__ Mlp2Smem sm;
  const int tid = threadIdx.x;
  const int lane = tid & 31;
  const int rowBase = (tid >> 5) * 16;
  const size_t rowG = (size_t)blockIdx.x * 64;

  loadWT(w1, 128, sm.w1T, tid);
  loadWT(w2, 64, sm.w2T, tid);
  if (tid < 64) { sm.b1s[tid] = b1[tid]; sm.b2s[tid] = b2[tid]; }
  for (int i = tid; i < 64 * 128; i += 128) {
    int row = i >> 7, c = i & 127;
    float v = (c < 64) ? X0[(rowG + row) * 64 + c] : X1[(rowG + row) * 64 + (c - 64)];
    sm.bufX[i] = (_Float16)v;
  }
  __syncthreads();
  waveGemm<128>(sm.bufX, 128, sm.w1T, sm.b1s, sm.bufH, 64, nullptr, 0, rowBase, lane);
  waveGemm<64>(sm.bufH, 64, sm.w2T, sm.b2s, nullptr, 0, out + rowG * 64, 64, rowBase, lane);
}

// ---------------------------------------------------------------------------
// Head helper kernels (fp32, tiny fraction of total FLOPs)
// ---------------------------------------------------------------------------
__global__ void __launch_bounds__(256)
head_gemm(const float* __restrict__ X0, const float* __restrict__ X1, int K,
          const float* __restrict__ W, const float* __restrict__ bias, int N,
          float* __restrict__ out, int relu_out)
{
  __shared__ float sW[128 * 64];
  __shared__ float sBias[64];
  const int tid = threadIdx.x;
  for (int i = tid; i < K * N; i += 256) sW[i] = W[i];
  if (tid < N) sBias[tid] = bias[tid];
  __syncthreads();
  const int gidx = blockIdx.x * 256 + tid;
  const int b = gidx / N, n = gidx % N;
  const float* xr0 = X0 + (size_t)b * 64;
  float s = sBias[n];
  for (int k = 0; k < 64; ++k) s += xr0[k] * sW[k * N + n];
  if (K == 128) {
    const float* xr1 = X1 + (size_t)b * 64;
    for (int k = 0; k < 64; ++k) s += xr1[k] * sW[(64 + k) * N + n];
  }
  if (relu_out) s = fmaxf(s, 0.f);
  out[gidx] = s;
}

__global__ void __launch_bounds__(256)
bn_stats(const float* __restrict__ X, int Bn, int C, float* __restrict__ stats)
{
  __shared__ float s1[256], s2[256];
  const int c = blockIdx.x, tid = threadIdx.x;
  float a = 0.f, q = 0.f;
  for (int i = tid; i < Bn; i += 256) {
    float v = X[(size_t)i * C + c];
    a += v; q += v * v;
  }
  s1[tid] = a; s2[tid] = q;
  __syncthreads();
  for (int off = 128; off > 0; off >>= 1) {
    if (tid < off) { s1[tid] += s1[tid + off]; s2[tid] += s2[tid + off]; }
    __syncthreads();
  }
  if (tid == 0) {
    float m = s1[0] / (float)Bn;
    float var = s2[0] / (float)Bn - m * m;
    stats[c * 2]     = m;
    stats[c * 2 + 1] = rsqrtf(var + 1e-5f);
  }
}

// out[b,n] = (relu_out?) relu( relu(bn(X[b,:])) @ W + bias )
__global__ void __launch_bounds__(256)
bn_relu_gemm(const float* __restrict__ X, const float* __restrict__ stats,
             const float* __restrict__ g, const float* __restrict__ bt,
             int K, const float* __restrict__ W, const float* __restrict__ bias, int N,
             float* __restrict__ out, int relu_out)
{
  __shared__ float sW[64 * 64];
  __shared__ float sScale[64], sShift[64], sBias[64];
  const int tid = threadIdx.x;
  for (int i = tid; i < K * N; i += 256) sW[i] = W[i];
  if (tid < K) {
    float m = stats[tid * 2], r = stats[tid * 2 + 1];
    sScale[tid] = r * g[tid];
    sShift[tid] = bt[tid] - m * r * g[tid];
  }
  if (tid < N) sBias[tid] = bias[tid];
  __syncthreads();
  const int gidx = blockIdx.x * 256 + tid;
  const int b = gidx / N, n = gidx % N;
  const float* xr = X + (size_t)b * K;
  float s = sBias[n];
  for (int k = 0; k < K; ++k) {
    float xv = fmaxf(xr[k] * sScale[k] + sShift[k], 0.f);
    s += xv * sW[k * N + n];
  }
  if (relu_out) s = fmaxf(s, 0.f);
  out[gidx] = s;
}

__global__ void __launch_bounds__(256)
final_kernel(const float* __restrict__ t4, const float* __restrict__ stats,
             const float* __restrict__ g, const float* __restrict__ bt,
             const float* __restrict__ w3, const float* __restrict__ b3,
             float* __restrict__ out, int Bn)
{
  const int b = blockIdx.x * 256 + threadIdx.x;
  if (b >= Bn) return;
  float s = b3[0];
  for (int n = 0; n < 16; ++n) {
    float xv = fmaxf((t4[(size_t)b * 16 + n] - stats[n * 2]) * stats[n * 2 + 1] * g[n] + bt[n], 0.f);
    s += xv * w3[n];
  }
  out[b] = s;
}

// ---------------------------------------------------------------------------
// d_in index map (setup_inputs order: 7 int inputs, then params in insertion order)
// ---------------------------------------------------------------------------
enum {
  IN_NODES_U = 0, IN_NODES_V, IN_HIST_U, IN_HIST_UR, IN_SOCIAL, IN_HIST_V, IN_HIST_VR,
  IN_EMBED_U, IN_EMBED_I, IN_EMBED_R,
  IN_GV1_W, IN_GV1_B, IN_GV2_W, IN_GV2_B,
  IN_ATTI1_W, IN_ATTI1_B, IN_ATTI2_W, IN_ATTI2_B, IN_ATTI3_W, IN_ATTI3_B,
  IN_ATTS1_W, IN_ATTS1_B, IN_ATTS2_W, IN_ATTS2_B, IN_ATTS3_W, IN_ATTS3_B,
  IN_UMLP1_W, IN_UMLP1_B, IN_UMLP2_W, IN_UMLP2_B,
  IN_GU1_W, IN_GU1_B, IN_GU2_W, IN_GU2_B,
  IN_ATTV1_W, IN_ATTV1_B, IN_ATTV2_W, IN_ATTV2_B, IN_ATTV3_W, IN_ATTV3_B,
  IN_WUR1_W, IN_WUR1_B, IN_WUR2_W, IN_WUR2_B,
  IN_WVR1_W, IN_WVR1_B, IN_WVR2_W, IN_WVR2_B,
  IN_WUV1_W, IN_WUV1_B, IN_WUV2_W, IN_WUV2_B, IN_WUV3_W, IN_WUV3_B,
  IN_BN1_G, IN_BN1_B, IN_BN2_G, IN_BN2_B, IN_BN3_G, IN_BN3_B, IN_BN4_G, IN_BN4_B
};

extern "C" void kernel_launch(void* const* d_in, const int* in_sizes, int n_in,
                              void* d_out, int out_size, void* d_ws, size_t ws_size,
                              hipStream_t stream)
{
  const int Bn = in_sizes[IN_NODES_U];   // 4096
  const size_t B64 = (size_t)Bn * 64;

  const int*   nodes_u  = (const int*)d_in[IN_NODES_U];
  const int*   nodes_v  = (const int*)d_in[IN_NODES_V];
  const int*   hist_u   = (const int*)d_in[IN_HIST_U];
  const int*   hist_ur  = (const int*)d_in[IN_HIST_UR];
  const int*   social   = (const int*)d_in[IN_SOCIAL];
  const int*   hist_v   = (const int*)d_in[IN_HIST_V];
  const int*   hist_vr  = (const int*)d_in[IN_HIST_VR];
  const float* embed_u  = (const float*)d_in[IN_EMBED_U];
  const float* embed_i  = (const float*)d_in[IN_EMBED_I];
  const float* embed_r  = (const float*)d_in[IN_EMBED_R];
  const float* F(int i) ; // (unused placeholder removed below)

  #define P(i) ((const float*)d_in[(i)])

  float* ws  = (float*)d_ws;
  float* hiI = ws;
  float* hiS = hiI + B64;
  float* zj0 = hiS + B64;
  float* hi  = zj0 + B64;
  float* t1  = hi  + B64;
  float* t2  = t1  + B64;
  float* tv  = t2  + B64;
  float* hif = tv  + B64;
  float* zjf = hif + B64;
  float* t3  = zjf + B64;
  float* t4  = t3  + B64;                       // Bn*16
  float* st1 = t4  + (size_t)Bn * 16;           // 4 stats blocks
  float* st2 = st1 + 128;
  float* st3 = st2 + 128;
  float* st4 = st3 + 128;

  // ---- user modeling: item-space and social-space aggregation ----
  branch_kernel<<<Bn, 128, 0, stream>>>(
      hist_u, hist_ur, embed_i, embed_r, nodes_u, embed_u,
      P(IN_GV1_W), P(IN_GV1_B), P(IN_GV2_W), P(IN_GV2_B),
      P(IN_ATTI1_W), P(IN_ATTI1_B), P(IN_ATTI2_W), P(IN_ATTI2_B),
      P(IN_ATTI3_W), P(IN_ATTI3_B), hiI);
  branch_kernel<<<Bn, 128, 0, stream>>>(
      social, nullptr, embed_u, nullptr, nodes_u, embed_u,
      nullptr, nullptr, nullptr, nullptr,
      P(IN_ATTS1_W), P(IN_ATTS1_B), P(IN_ATTS2_W), P(IN_ATTS2_B),
      P(IN_ATTS3_W), P(IN_ATTS3_B), hiS);
  // ---- item modeling ----
  branch_kernel<<<Bn, 128, 0, stream>>>(
      hist_v, hist_vr, embed_u, embed_r, nodes_v, embed_i,
      P(IN_GU1_W), P(IN_GU1_B), P(IN_GU2_W), P(IN_GU2_B),
      P(IN_ATTV1_W), P(IN_ATTV1_B), P(IN_ATTV2_W), P(IN_ATTV2_B),
      P(IN_ATTV3_W), P(IN_ATTV3_B), zj0);

  // hi = umlp(concat(hiI, hiS))
  mlp2_kernel<<<Bn / 64, 128, 0, stream>>>(
      hiI, hiS, P(IN_UMLP1_W), P(IN_UMLP1_B), P(IN_UMLP2_W), P(IN_UMLP2_B), hi);

  // ---- rating head ----
  head_gemm<<<Bn * 64 / 256, 256, 0, stream>>>(hi,  nullptr, 64, P(IN_WUR1_W), P(IN_WUR1_B), 64, t1, 0);
  head_gemm<<<Bn * 64 / 256, 256, 0, stream>>>(zj0, nullptr, 64, P(IN_WVR1_W), P(IN_WVR1_B), 64, t2, 0);
  bn_stats<<<64, 256, 0, stream>>>(t1, Bn, 64, st1);
  bn_stats<<<64, 256, 0, stream>>>(t2, Bn, 64, st2);
  bn_relu_gemm<<<Bn * 64 / 256, 256, 0, stream>>>(t1, st1, P(IN_BN1_G), P(IN_BN1_B), 64,
                                                  P(IN_WUR2_W), P(IN_WUR2_B), 64, hif, 0);
  bn_relu_gemm<<<Bn * 64 / 256, 256, 0, stream>>>(t2, st2, P(IN_BN2_G), P(IN_BN2_B), 64,
                                                  P(IN_WVR1_W), P(IN_WVR1_B), 64, tv, 1);
  head_gemm<<<Bn * 64 / 256, 256, 0, stream>>>(tv, nullptr, 64, P(IN_WVR2_W), P(IN_WVR2_B), 64, zjf, 0);
  head_gemm<<<Bn * 64 / 256, 256, 0, stream>>>(hif, zjf, 128, P(IN_WUV1_W), P(IN_WUV1_B), 64, t3, 0);
  bn_stats<<<64, 256, 0, stream>>>(t3, Bn, 64, st3);
  bn_relu_gemm<<<Bn * 16 / 256, 256, 0, stream>>>(t3, st3, P(IN_BN3_G), P(IN_BN3_B), 64,
                                                  P(IN_WUV2_W), P(IN_WUV2_B), 16, t4, 0);
  bn_stats<<<16, 256, 0, stream>>>(t4, Bn, 16, st4);
  final_kernel<<<(Bn + 255) / 256, 256, 0, stream>>>(t4, st4, P(IN_BN4_G), P(IN_BN4_B),
                                                     P(IN_WUV3_W), P(IN_WUV3_B),
                                                     (float*)d_out, Bn);
  #undef P
}
